// GraphAttentionLayer_43834436223481
// MI455X (gfx1250) — matrix operand order repaired and checked by reading, not compiled, not run on hardware
//
#include <hip/hip_runtime.h>
#include <hip/hip_bf16.h>

// GAT fused kernel for MI455X (gfx1250, wave32, WMMA).
// B=8, N=2048, H=4, DIN=128, DOUT=64, ALPHA=0.2
#define GB    8
#define GN    2048
#define GH    4
#define GDIN  128
#define GDOUT 64
#define GALPHA 0.2f

typedef __attribute__((ext_vector_type(16))) _Float16 v16h;
typedef __attribute__((ext_vector_type(8)))  _Float16 v8h;
typedef __attribute__((ext_vector_type(8)))  float    v8f;

// ---------------------------------------------------------------------------
// Kernel 0: WhT[h,c,k] = (f16) W[h,k,c]  (transpose so WMMA B-fragments are
// contiguous). Tiny: 32K elements, one block.
// ---------------------------------------------------------------------------
__global__ __launch_bounds__(256) void gat_wprep(const float* __restrict__ W,
                                                 _Float16* __restrict__ WhT) {
    const int hc = threadIdx.x;          // h*64 + c
    const int h = hc >> 6, c = hc & 63;
#pragma unroll 8
    for (int k = 0; k < GDIN; ++k)
        WhT[(size_t)hc * GDIN + k] = (_Float16)W[((size_t)h * GDIN + k) * GDOUT + c];
}

// ---------------------------------------------------------------------------
// Kernel 1: xtT[b,h,c,n] = (f16) sum_k x[b,n,k] * W[h,k,c]
// One wave per (b,h, 16-node tile). 16 WMMAs (K=128 in 4 chunks, 4 col tiles).
// Stored transposed so kernel 4's B-fragments are contiguous in memory.
// ---------------------------------------------------------------------------
__global__ __launch_bounds__(32) void gat_proj(const float* __restrict__ x,
                                               const _Float16* __restrict__ WhT,
                                               _Float16* __restrict__ xtT) {
    const int bh   = blockIdx.x >> 7;          // 0..31  (b*4 + h)
    const int tile = blockIdx.x & 127;         // 0..127
    const int b    = bh >> 2;
    const int h    = bh & 3;
    const int i0   = tile << 4;

    const int lane  = threadIdx.x;
    const int row16 = lane & 15;
    const int hi    = lane >> 4;

    v8f acc[4];
#pragma unroll
    for (int ct = 0; ct < 4; ++ct) acc[ct] = (v8f)0.0f;

    const float* xrow = x + ((size_t)b * GN + i0 + row16) * GDIN;

#pragma unroll
    for (int kc = 0; kc < 4; ++kc) {
        // A fragment: 16x32 f16, lane holds row (l&15), K contiguous per lane group
        v16h A;
#pragma unroll
        for (int v = 0; v < 8; ++v) {
            const int k = kc * 32 + (v << 1) + (hi << 4);
            A[2 * v]     = (_Float16)xrow[k];
            A[2 * v + 1] = (_Float16)xrow[k + 1];
        }
#pragma unroll
        for (int ct = 0; ct < 4; ++ct) {
            // B fragment: 32x16 f16, lane = col, K contiguous per lane group
            const int col = ct * 16 + row16;
            const int kb  = kc * 32 + (hi << 4);
            const _Float16* wp = WhT + ((size_t)(h * GDOUT + col)) * GDIN + kb;
            const v8h w0 = *(const v8h*)wp;
            const v8h w1 = *(const v8h*)(wp + 8);
            const v16h Bf = __builtin_shufflevector(
                w0, w1, 0, 1, 2, 3, 4, 5, 6, 7, 8, 9, 10, 11, 12, 13, 14, 15);
            acc[ct] = __builtin_amdgcn_wmma_f32_16x16x32_f16(
                false, A, false, Bf, (short)0, acc[ct], false, false);
        }
    }

    // C layout: lane = col (l&15), VGPR v = row (v + hi*8). Rows are
    // contiguous in xtT's n-dimension -> one 16B store per col tile.
#pragma unroll
    for (int ct = 0; ct < 4; ++ct) {
        const int c = ct * 16 + row16;
        v8h o;
#pragma unroll
        for (int v = 0; v < 8; ++v) o[v] = (_Float16)acc[ct][v];
        *(v8h*)(xtT + ((size_t)(bh * GDOUT + c)) * GN + i0 + hi * 8) = o;
    }
}

// ---------------------------------------------------------------------------
// Kernel 2: s_i[bh,n] = xt[n,:] . a_l[h],  s_j[bh,n] = xt[n,:] . a_r[h]
// Coalesced over n (xtT is [c][n]).
// ---------------------------------------------------------------------------
__global__ __launch_bounds__(256) void gat_scores(const _Float16* __restrict__ xtT,
                                                  const float* __restrict__ attn,
                                                  float* __restrict__ s_i,
                                                  float* __restrict__ s_j) {
    const int bh = blockIdx.x >> 3;                     // 0..31
    const int n  = ((blockIdx.x & 7) << 8) + threadIdx.x;
    const int h  = bh & 3;
    float si = 0.f, sj = 0.f;
#pragma unroll 8
    for (int c = 0; c < GDOUT; ++c) {
        const float v = (float)xtT[((size_t)(bh * GDOUT + c)) * GN + n];
        si += v * attn[h * (2 * GDOUT) + c];
        sj += v * attn[h * (2 * GDOUT) + GDOUT + c];
    }
    s_i[bh * GN + n] = si;
    s_j[bh * GN + n] = sj;
}

// ---------------------------------------------------------------------------
// Kernel 3: smax[bh] = max_n s_j[bh,n]  (unmasked upper bound for row maxima;
// leaky_relu is monotone so m_row = leaky(s_i + smax) >= true masked row max,
// and the slack is O(score range) so exp never underflows f32).
// ---------------------------------------------------------------------------
__global__ __launch_bounds__(256) void gat_smax(const float* __restrict__ s_j,
                                                float* __restrict__ smax) {
    __shared__ float red[256];
    const int bh = blockIdx.x;
    float m = -__builtin_inff();
    for (int n = threadIdx.x; n < GN; n += 256)
        m = fmaxf(m, s_j[bh * GN + n]);
    red[threadIdx.x] = m;
    __syncthreads();
#pragma unroll
    for (int s = 128; s > 0; s >>= 1) {
        if ((int)threadIdx.x < s)
            red[threadIdx.x] = fmaxf(red[threadIdx.x], red[threadIdx.x + s]);
        __syncthreads();
    }
    if (threadIdx.x == 0) smax[bh] = red[0];
}

// ---------------------------------------------------------------------------
// Kernel 4: streaming masked softmax + attn @ xt, static row max.
// Block = 128 threads = 4 waves; wave w = head w; block covers (b, 32 rows):
// each wave owns TWO 16-row A tiles sharing one set of B fragments (halves
// xtT L2 traffic). adj/sj tiles double-buffered in LDS -> 1 barrier/step and
// next-tile HBM loads overlap current-tile exp+WMMA work.
// ---------------------------------------------------------------------------
__global__ __launch_bounds__(128) void gat_attn(const int* __restrict__ adj,
                                                const _Float16* __restrict__ xtT,
                                                const float* __restrict__ s_i,
                                                const float* __restrict__ s_j,
                                                const float* __restrict__ smax,
                                                float* __restrict__ out) {
    __shared__ __align__(16) int adjT[2][32 * 32];
    __shared__ float sjs[2][GH][32];
    __shared__ float rowtmp[GH][32];

    const int b  = blockIdx.x >> 6;            // grid = 8 * 64
    const int i0 = (blockIdx.x & 63) << 5;     // 32-row block
    const int h  = threadIdx.x >> 5;
    const int lane  = threadIdx.x & 31;
    const int bh    = b * GH + h;
    const int row16 = lane & 15;
    const int hi    = lane >> 4;

    const float smx = smax[bh];
    float si_r[2], m_r[2], psum[2];
    v8f acc[2][4];
#pragma unroll
    for (int rt = 0; rt < 2; ++rt) {
        si_r[rt] = s_i[bh * GN + i0 + rt * 16 + row16];
        const float m0 = si_r[rt] + smx;          // static row-max bound
        m_r[rt] = m0 > 0.f ? m0 : GALPHA * m0;
        psum[rt] = 0.f;
#pragma unroll
        for (int ct = 0; ct < 4; ++ct) acc[rt][ct] = (v8f)0.0f;
    }

    const size_t adjbase = (size_t)b * GN * GN + (size_t)i0 * GN;
    const int t  = threadIdx.x;
    const int sr = t >> 2;                 // stager: row 0..31
    const int sc = (t & 3) << 3;           // stager: col 0,8,16,24

    // stage tile 0 into buffer 0
    {
        const int j0 = 0;
        *(int4*)&adjT[0][sr * 32 + sc] =
            *(const int4*)(adj + adjbase + (size_t)sr * GN + j0 + sc);
        *(int4*)&adjT[0][sr * 32 + sc + 4] =
            *(const int4*)(adj + adjbase + (size_t)sr * GN + j0 + sc + 4);
        sjs[0][h][lane] = s_j[bh * GN + j0 + lane];
    }
    __syncthreads();

    for (int jb = 0; jb < 64; ++jb) {
        const int cur = jb & 1;
        const int j0  = jb * 32;

        // prefetch next tile into the other buffer (overlaps compute)
        if (jb + 1 < 64) {
            const int jn = j0 + 32;
            *(int4*)&adjT[cur ^ 1][sr * 32 + sc] =
                *(const int4*)(adj + adjbase + (size_t)sr * GN + jn + sc);
            *(int4*)&adjT[cur ^ 1][sr * 32 + sc + 4] =
                *(const int4*)(adj + adjbase + (size_t)sr * GN + jn + sc + 4);
            sjs[cur ^ 1][h][lane] = s_j[bh * GN + jn + lane];
        }

        // P fragments (K contiguous per lane group) for both 16-row tiles
        v16h P[2];
#pragma unroll
        for (int rt = 0; rt < 2; ++rt) {
            const int arow = (rt * 16 + row16) * 32;
#pragma unroll
            for (int v = 0; v < 8; ++v) {
                const int kb = (v << 1) + (hi << 4);
#pragma unroll
                for (int d = 0; d < 2; ++d) {
                    const int kk = kb + d;
                    float e = si_r[rt] + sjs[cur][h][kk];
                    e = e > 0.f ? e : GALPHA * e;
                    const float pe =
                        (adjT[cur][arow + kk] != 0) ? __expf(e - m_r[rt]) : 0.f;
                    psum[rt] += pe;
                    P[rt][2 * v + d] = (_Float16)pe;
                }
            }
        }

        // one B fragment feeds both row tiles (2x FLOPs per loaded byte)
        const int kb = j0 + hi * 16;
#pragma unroll
        for (int ct = 0; ct < 4; ++ct) {
            const int c = ct * 16 + row16;
            const _Float16* bp = xtT + ((size_t)(bh * GDOUT + c)) * GN + kb;
            const v8h b0 = *(const v8h*)bp;
            const v8h b1 = *(const v8h*)(bp + 8);
            const v16h Bf = __builtin_shufflevector(
                b0, b1, 0, 1, 2, 3, 4, 5, 6, 7, 8, 9, 10, 11, 12, 13, 14, 15);
            acc[0][ct] = __builtin_amdgcn_wmma_f32_16x16x32_f16(
                false, P[0], false, Bf, (short)0, acc[0][ct], false, false);
            acc[1][ct] = __builtin_amdgcn_wmma_f32_16x16x32_f16(
                false, P[1], false, Bf, (short)0, acc[1][ct], false, false);
        }
        __syncthreads();   // buffer cur consumed; cur^1 fully staged
    }

    // lanes l and l^16 hold the same row: combine partial sums once
#pragma unroll
    for (int rt = 0; rt < 2; ++rt) psum[rt] += __shfl_xor(psum[rt], 16);

    // epilogue: divide by row sums, write out[b, n, h*64 + c]
    if (lane < 16) {
        rowtmp[h][lane]      = psum[0];
        rowtmp[h][lane + 16] = psum[1];
    }
#pragma unroll
    for (int rt = 0; rt < 2; ++rt) {
        float linv[8];
#pragma unroll
        for (int v = 0; v < 8; ++v) {
            const float s = rowtmp[h][rt * 16 + hi * 8 + v];
            linv[v] = (s > 0.f) ? (1.f / s) : 0.f;
        }
#pragma unroll
        for (int ct = 0; ct < 4; ++ct) {
            const int c = ct * 16 + row16;
#pragma unroll
            for (int v = 0; v < 8; ++v) {
                out[((size_t)b * GN + i0 + rt * 16 + hi * 8 + v) * (GH * GDOUT) +
                    h * GDOUT + c] = acc[rt][ct][v] * linv[v];
            }
        }
    }
}

// ---------------------------------------------------------------------------
extern "C" void kernel_launch(void* const* d_in, const int* in_sizes, int n_in,
                              void* d_out, int out_size, void* d_ws, size_t ws_size,
                              hipStream_t stream) {
    (void)in_sizes; (void)n_in; (void)out_size; (void)ws_size;
    const float* x    = (const float*)d_in[0];   // (B, N, DIN) f32
    const int*   adj  = (const int*)d_in[1];     // (B, N, N) i32
    const float* W    = (const float*)d_in[2];   // (H, DIN, DOUT) f32
    const float* attn = (const float*)d_in[3];   // (H, 2*DOUT, 1) f32
    float* out = (float*)d_out;                  // (B, N, H*DOUT) f32

    // workspace layout
    char* ws = (char*)d_ws;
    _Float16* xtT = (_Float16*)ws;                               // 8 MiB
    size_t off = (size_t)GB * GH * GDOUT * GN * sizeof(_Float16);
    float* s_i = (float*)(ws + off);
    off += (size_t)GB * GH * GN * sizeof(float);
    float* s_j = (float*)(ws + off);
    off += (size_t)GB * GH * GN * sizeof(float);
    _Float16* WhT = (_Float16*)(ws + off);                       // 64 KiB
    off += (size_t)GH * GDOUT * GDIN * sizeof(_Float16);
    float* smax = (float*)(ws + off);                            // 32 floats

    gat_wprep <<<1, 256, 0, stream>>>(W, WhT);
    gat_proj  <<<GB * GH * (GN / 16), 32,  0, stream>>>(x, WhT, xtT);
    gat_scores<<<GB * GH * (GN / 256), 256, 0, stream>>>(xtT, attn, s_i, s_j);
    gat_smax  <<<GB * GH, 256, 0, stream>>>(s_j, smax);
    gat_attn  <<<GB * (GN / 32), 128, 0, stream>>>(adj, xtT, s_i, s_j, smax, out);
}
